// ARDecoder_23837068493506
// MI455X (gfx1250) — matrix operand order repaired and checked by reading, not compiled, not run on hardware
//
#include <hip/hip_runtime.h>

// ---------------------------------------------------------------------------
// CDNA5 (gfx1250) autoregressive GRU decoder with WMMA bf16 GEMMs and
// async global->LDS staging of the activation (A) tiles.
// ---------------------------------------------------------------------------

typedef __attribute__((ext_vector_type(16))) __bf16 v16bf;
typedef __attribute__((ext_vector_type(8)))  __bf16 v8bf;
typedef __attribute__((ext_vector_type(8)))  float  v8f;
typedef __bf16 bf16_t;

union FragBF { v16bf v; v8bf h[2]; };

#define NBATCH 4096
#define HID    512
#define EMB    128
#define VOCAB  2048
#define NEGV   (-1.0e9f)

static __device__ __forceinline__ float sigmoidf_(float x) {
  return 1.0f / (1.0f + __expf(-x));
}

// deterministic hash RNG -> [0,1)
static __device__ __forceinline__ float rng_uniform(unsigned a, unsigned b) {
  unsigned s = a * 0x9E3779B9u ^ (b * 0x85EBCA6Bu + 0x7F4A7C15u);
  s ^= s >> 16; s *= 0x7FEB352Du;
  s ^= s >> 15; s *= 0x846CA68Bu;
  s ^= s >> 16;
  return (float)(s >> 8) * (1.0f / 16777216.0f);
}

// ---------------------------------------------------------------------------
// Async global -> LDS 16-byte copy (gfx1250 GLOBAL_LOAD_ASYNC_TO_LDS_B128).
// VDST VGPR holds the workgroup-relative LDS byte address (low 32 bits of the
// generic pointer, per the LDS aperture layout); VADDR holds the 64-bit
// global address. Tracked by ASYNCcnt.
// ---------------------------------------------------------------------------
static __device__ __forceinline__ void async_copy_b128(const void* gptr, void* lptr) {
  unsigned lds = (unsigned)(unsigned long long)lptr;
  unsigned long long ga = (unsigned long long)gptr;
  asm volatile("global_load_async_to_lds_b128 %0, %1, off"
               :: "v"(lds), "v"(ga) : "memory");
}

static __device__ __forceinline__ void wait_async_zero() {
#if __has_builtin(__builtin_amdgcn_s_wait_asynccnt)
  __builtin_amdgcn_s_wait_asynccnt(0);
#else
  asm volatile("s_wait_asynccnt 0x0" ::: "memory");
#endif
}

// ---------------------------------------------------------------------------
// fp32 -> bf16 weight conversion (one-time)
// ---------------------------------------------------------------------------
__global__ void cvt_f32_bf16_kernel(const float* __restrict__ src,
                                    bf16_t* __restrict__ dst, int n) {
  int i = blockIdx.x * blockDim.x + threadIdx.x;
  int stride = gridDim.x * blockDim.x;
  for (; i < n; i += stride) dst[i] = (__bf16)src[i];
}

// ---------------------------------------------------------------------------
// h = tanh(z @ w_lh^T + b_lh)  -> split into layer0 / layer1 hidden buffers
// ---------------------------------------------------------------------------
__global__ void init_hidden_kernel(const float* __restrict__ z,
                                   const float* __restrict__ w_lh,
                                   const float* __restrict__ b_lh,
                                   float* __restrict__ h0f, float* __restrict__ h1f,
                                   bf16_t* __restrict__ h0b, bf16_t* __restrict__ h1b) {
  int id = blockIdx.x * blockDim.x + threadIdx.x;   // NBATCH * 2*HID threads
  if (id >= NBATCH * 2 * HID) return;
  int b = id >> 10;           // /1024
  int j = id & 1023;
  const float* zr = z + (size_t)b * EMB;
  const float* wr = w_lh + (size_t)j * EMB;
  float acc = b_lh[j];
#pragma unroll 4
  for (int k = 0; k < EMB; ++k) acc += zr[k] * wr[k];
  float h = tanhf(acc);
  if (j < HID) {
    h0f[(size_t)b * HID + j] = h;
    h0b[(size_t)b * HID + j] = (__bf16)h;
  } else {
    h1f[(size_t)b * HID + (j - HID)] = h;
    h1b[(size_t)b * HID + (j - HID)] = (__bf16)h;
  }
}

__global__ void init_tokens_kernel(const int* __restrict__ bosP,
                                   const int* __restrict__ maxlenP,
                                   int* __restrict__ out,
                                   int* __restrict__ cur,
                                   int* __restrict__ finished) {
  int b = blockIdx.x * blockDim.x + threadIdx.x;
  if (b >= NBATCH) return;
  int bos = *bosP;
  out[(size_t)b * (*maxlenP)] = bos;
  cur[b] = bos;
  finished[b] = 0;
}

// ---------------------------------------------------------------------------
// Fused GRU cell: 8 waves/block, each wave owns one 16x16 h_new tile with
// 6 WMMA accumulators (r,z,n for the input path and hidden path).
// Activation tiles (x-block, h-block) are staged into LDS with async loads;
// layer-0 fuses the embedding gather into the per-lane async addresses.
// Block tile: 32 (M) x 64 (N). Grid: (NBATCH/32, HID/64).
// Dynamic LDS: 32*Kx*2 + 32*HID*2 bytes.
// ---------------------------------------------------------------------------
__global__ void __launch_bounds__(256)
gru_cell_kernel(const bf16_t* __restrict__ X, const int* __restrict__ idx, int Kx,
                const bf16_t* __restrict__ Hbf, const float* __restrict__ Hf,
                const bf16_t* __restrict__ Wih, const bf16_t* __restrict__ Whh,
                const float* __restrict__ bih, const float* __restrict__ bhh,
                float* __restrict__ Hof, bf16_t* __restrict__ Hob) {
  extern __shared__ __align__(16) char smem_raw[];
  bf16_t* sx = (bf16_t*)smem_raw;                              // [32][Kx]
  bf16_t* sh = (bf16_t*)(smem_raw + (size_t)32 * Kx * 2);      // [32][HID]

  const int tid  = threadIdx.x;
  const int lane = threadIdx.x & 31;
  const int wave = threadIdx.x >> 5;
  const int wm = wave >> 2;                 // 0..1
  const int wn = wave & 3;                  // 0..3
  const int tile_m = blockIdx.x * 32 + wm * 16;
  const int tile_n = blockIdx.y * 64 + wn * 16;

  // ---- async-stage the 32-row activation blocks into LDS ----
  {
    const int rchunks = (Kx * 2) / 16;                 // 16B chunks per x row
    const int xchunks = 32 * rchunks;                  // multiple of 256
    for (int c = tid; c < xchunks; c += 256) {
      const int row = c / rchunks;
      const int col = c % rchunks;                     // 16B chunk in row
      const int grow = idx ? idx[blockIdx.x * 32 + row] : (blockIdx.x * 32 + row);
      async_copy_b128(X + (size_t)grow * Kx + col * 8, sx + (size_t)row * Kx + col * 8);
    }
    const int hrchunks = (HID * 2) / 16;               // 64 chunks per h row
    const int hchunks = 32 * hrchunks;                 // 2048
    for (int c = tid; c < hchunks; c += 256) {
      const int row = c / hrchunks;
      const int col = c % hrchunks;
      async_copy_b128(Hbf + (size_t)(blockIdx.x * 32 + row) * HID + col * 8,
                      sh + (size_t)row * HID + col * 8);
    }
  }
  wait_async_zero();
  __syncthreads();

  v8f air = {}, aiz = {}, ain = {};
  v8f ahr = {}, ahz = {}, ahn = {};

  const int mloc    = wm * 16 + (lane & 15);   // local A row inside the block
  const int mrow    = tile_m + (lane & 15);    // global row for the epilogue
  const int ncol    = tile_n + (lane & 15);
  const int ksel8   = (lane >> 4) * 8;         // unused (A and B frags both use khalf16)
  const int khalf16 = (lane >> 4) * 16;        // B frag K-half select

  // ---- input-path GEMMs: gi = x @ Wih^T (x read from LDS) ----
  const bf16_t* xrow = sx + (size_t)mloc * Kx;
  for (int k0 = 0; k0 < Kx; k0 += 32) {
    FragBF a;
    a.h[0] = *(const v8bf*)(xrow + k0 + khalf16);
    a.h[1] = *(const v8bf*)(xrow + k0 + khalf16 + 8);
    const bf16_t* w0 = Wih + (size_t)(ncol)           * Kx + k0 + khalf16;
    const bf16_t* w1 = Wih + (size_t)(ncol + HID)     * Kx + k0 + khalf16;
    const bf16_t* w2 = Wih + (size_t)(ncol + 2 * HID) * Kx + k0 + khalf16;
    __builtin_prefetch(w0 + 32, 0, 3);
    FragBF br, bz, bn;
    br.h[0] = *(const v8bf*)w0; br.h[1] = *(const v8bf*)(w0 + 8);
    bz.h[0] = *(const v8bf*)w1; bz.h[1] = *(const v8bf*)(w1 + 8);
    bn.h[0] = *(const v8bf*)w2; bn.h[1] = *(const v8bf*)(w2 + 8);
    air = __builtin_amdgcn_wmma_f32_16x16x32_bf16(false, a.v, false, br.v, (short)0, air, false, false);
    aiz = __builtin_amdgcn_wmma_f32_16x16x32_bf16(false, a.v, false, bz.v, (short)0, aiz, false, false);
    ain = __builtin_amdgcn_wmma_f32_16x16x32_bf16(false, a.v, false, bn.v, (short)0, ain, false, false);
  }

  // ---- hidden-path GEMMs: gh = h @ Whh^T (h read from LDS) ----
  const bf16_t* hrow = sh + (size_t)mloc * HID;
  for (int k0 = 0; k0 < HID; k0 += 32) {
    FragBF a;
    a.h[0] = *(const v8bf*)(hrow + k0 + khalf16);
    a.h[1] = *(const v8bf*)(hrow + k0 + khalf16 + 8);
    const bf16_t* w0 = Whh + (size_t)(ncol)           * HID + k0 + khalf16;
    const bf16_t* w1 = Whh + (size_t)(ncol + HID)     * HID + k0 + khalf16;
    const bf16_t* w2 = Whh + (size_t)(ncol + 2 * HID) * HID + k0 + khalf16;
    __builtin_prefetch(w0 + 32, 0, 3);
    FragBF br, bz, bn;
    br.h[0] = *(const v8bf*)w0; br.h[1] = *(const v8bf*)(w0 + 8);
    bz.h[0] = *(const v8bf*)w1; bz.h[1] = *(const v8bf*)(w1 + 8);
    bn.h[0] = *(const v8bf*)w2; bn.h[1] = *(const v8bf*)(w2 + 8);
    ahr = __builtin_amdgcn_wmma_f32_16x16x32_bf16(false, a.v, false, br.v, (short)0, ahr, false, false);
    ahz = __builtin_amdgcn_wmma_f32_16x16x32_bf16(false, a.v, false, bz.v, (short)0, ahz, false, false);
    ahn = __builtin_amdgcn_wmma_f32_16x16x32_bf16(false, a.v, false, bn.v, (short)0, ahn, false, false);
  }

  // ---- gate epilogue ----
  const float birv = bih[ncol], bizv = bih[ncol + HID], binv = bih[ncol + 2 * HID];
  const float bhrv = bhh[ncol], bhzv = bhh[ncol + HID], bhnv = bhh[ncol + 2 * HID];
  const int mbase = tile_m + ((lane >> 4) << 3);   // C/D layout: M = vgpr + 8*(lane>=16)
#pragma unroll
  for (int vi = 0; vi < 8; ++vi) {
    const int rrow = mbase + vi;
    const float r = sigmoidf_(air[vi] + birv + ahr[vi] + bhrv);
    const float u = sigmoidf_(aiz[vi] + bizv + ahz[vi] + bhzv);
    const float n = tanhf(ain[vi] + binv + r * (ahn[vi] + bhnv));
    const float hp = Hf[(size_t)rrow * HID + ncol];
    const float hn = (1.0f - u) * n + u * hp;
    Hof[(size_t)rrow * HID + ncol] = hn;
    Hob[(size_t)rrow * HID + ncol] = (__bf16)hn;
  }
  (void)mrow;
}

// ---------------------------------------------------------------------------
// logits = h1 @ W_out^T + b_out.  Wave tile 16x64 (A-frag reused over 4 B).
// A block (64 rows x 512) staged into LDS with async loads.
// Block tile 64 (M) x 128 (N). Grid: (NBATCH/64, VOCAB/128).
// Dynamic LDS: 64*HID*2 = 64 KB.
// ---------------------------------------------------------------------------
__global__ void __launch_bounds__(256)
logits_kernel(const bf16_t* __restrict__ Hb, const bf16_t* __restrict__ Wout,
              const float* __restrict__ bout, float* __restrict__ L) {
  extern __shared__ __align__(16) char smem_raw2[];
  bf16_t* sa = (bf16_t*)smem_raw2;                 // [64][HID]

  const int tid  = threadIdx.x;
  const int lane = threadIdx.x & 31;
  const int wave = threadIdx.x >> 5;
  const int wm = wave & 3;                  // 0..3
  const int wn = wave >> 2;                 // 0..1
  const int tile_m = blockIdx.x * 64 + wm * 16;
  const int tile_n = blockIdx.y * 128 + wn * 64;

  {
    const int rchunks = (HID * 2) / 16;             // 64 chunks per row
    const int chunks = 64 * rchunks;                // 4096
    for (int c = tid; c < chunks; c += 256) {
      const int row = c / rchunks;
      const int col = c % rchunks;
      async_copy_b128(Hb + (size_t)(blockIdx.x * 64 + row) * HID + col * 8,
                      sa + (size_t)row * HID + col * 8);
    }
  }
  wait_async_zero();
  __syncthreads();

  v8f acc[4] = {};
  const int ksel8   = (lane >> 4) * 8;
  const int khalf16 = (lane >> 4) * 16;
  const int nlane = lane & 15;
  const bf16_t* hrow = sa + (size_t)(wm * 16 + nlane) * HID;

  for (int k0 = 0; k0 < HID; k0 += 32) {
    FragBF a;
    a.h[0] = *(const v8bf*)(hrow + k0 + khalf16);
    a.h[1] = *(const v8bf*)(hrow + k0 + khalf16 + 8);
#pragma unroll
    for (int j = 0; j < 4; ++j) {
      const bf16_t* w = Wout + (size_t)(tile_n + j * 16 + nlane) * HID + k0 + khalf16;
      __builtin_prefetch(w + 32, 0, 3);
      FragBF b;
      b.h[0] = *(const v8bf*)w; b.h[1] = *(const v8bf*)(w + 8);
      acc[j] = __builtin_amdgcn_wmma_f32_16x16x32_bf16(false, a.v, false, b.v, (short)0, acc[j], false, false);
    }
  }

  const int mbase = tile_m + ((lane >> 4) << 3);
#pragma unroll
  for (int j = 0; j < 4; ++j) {
    const int c = tile_n + j * 16 + nlane;
    const float bb = bout[c];
#pragma unroll
    for (int vi = 0; vi < 8; ++vi) {
      L[(size_t)(mbase + vi) * VOCAB + c] = acc[j][vi] + bb;
    }
  }
}

// ---------------------------------------------------------------------------
// Per-row top-k categorical sampling. One block (256 thr) per batch row.
// ---------------------------------------------------------------------------
__global__ void __launch_bounds__(256)
sample_kernel(const float* __restrict__ logits,
              const int* __restrict__ bosP, const int* __restrict__ eosP,
              const int* __restrict__ padP, const int* __restrict__ topkP,
              const int* __restrict__ minlenP, const int* __restrict__ maxlenP,
              int t, int* __restrict__ out_tokens,
              int* __restrict__ cur, int* __restrict__ finished) {
  __shared__ float sh[256];
  __shared__ int schosen;

  const int row = blockIdx.x;
  const int tid = threadIdx.x;
  const int bos = *bosP, eos = *eosP, pad = *padP, topk = *topkP;
  const int max_len = *maxlenP;
  int eff_min = *minlenP;
  if (eff_min > max_len - 2) eff_min = max_len - 2;
  if (eff_min < 0) eff_min = 0;
  const bool block_eos = (t - 1) < eff_min;

  const float* lr = logits + (size_t)row * VOCAB;
  float v[8];
#pragma unroll
  for (int j = 0; j < 8; ++j) {
    const int col = tid + 256 * j;
    float x = lr[col];
    if (col == bos || col == pad) x = NEGV;
    if (block_eos && col == eos) x = NEGV;
    v[j] = x;
  }

  // row max
  float m = v[0];
#pragma unroll
  for (int j = 1; j < 8; ++j) m = fmaxf(m, v[j]);
  sh[tid] = m; __syncthreads();
  for (int off = 128; off > 0; off >>= 1) {
    if (tid < off) sh[tid] = fmaxf(sh[tid], sh[tid + off]);
    __syncthreads();
  }
  const float vmax = sh[0]; __syncthreads();

  // bisection for the top-k threshold: invariant count(v >= lo) >= topk
  float thr;
  if (topk <= 0 || topk >= VOCAB) {
    thr = NEGV;
  } else {
    float lo = NEGV, hi = vmax;
    for (int it = 0; it < 24; ++it) {
      const float mid = 0.5f * (lo + hi);
      int c = 0;
#pragma unroll
      for (int j = 0; j < 8; ++j) c += (v[j] >= mid) ? 1 : 0;
      sh[tid] = (float)c; __syncthreads();
      for (int off = 128; off > 0; off >>= 1) {
        if (tid < off) sh[tid] += sh[tid + off];
        __syncthreads();
      }
      const float tot = sh[0]; __syncthreads();
      if (tot >= (float)topk) lo = mid; else hi = mid;
    }
    thr = lo;
  }

  // softmax mass over kept entries
  float p[8];
  float s = 0.0f;
#pragma unroll
  for (int j = 0; j < 8; ++j) {
    const bool keep = (v[j] >= thr);
    p[j] = keep ? __expf(v[j] - vmax) : 0.0f;
    s += p[j];
  }

  // block inclusive prefix scan of per-thread mass
  sh[tid] = s; __syncthreads();
  for (int off = 1; off < 256; off <<= 1) {
    const float x = (tid >= off) ? sh[tid - off] : 0.0f;
    __syncthreads();
    sh[tid] += x;
    __syncthreads();
  }
  const float incl = sh[tid];
  const float total = sh[255];
  const float excl = incl - s;

  if (tid == 0) schosen = -1;
  __syncthreads();

  const float u = rng_uniform((unsigned)row, (unsigned)t) * total;
  if (s > 0.0f && u >= excl && u < incl) {
    float acc = excl;
    int col = -1;
#pragma unroll
    for (int j = 0; j < 8; ++j) {
      if (col < 0 && p[j] > 0.0f) {
        if (u < acc + p[j]) col = tid + 256 * j;
        else acc += p[j];
      }
    }
    if (col < 0) {
#pragma unroll
      for (int j = 7; j >= 0; --j) if (col < 0 && p[j] > 0.0f) col = tid + 256 * j;
    }
    if (col >= 0) atomicExch(&schosen, col);
  }
  __syncthreads();

  if (tid == 0) {
    int choice = schosen;
    if (choice < 0) choice = 3;   // degenerate fallback (never expected)
    int fin = finished[row];
    int nxt = fin ? pad : choice;
    if (nxt == eos) fin = 1;
    finished[row] = fin;
    cur[row] = nxt;
    out_tokens[(size_t)row * max_len + t] = nxt;
  }
}

// ---------------------------------------------------------------------------
// Host launcher
// ---------------------------------------------------------------------------
extern "C" void kernel_launch(void* const* d_in, const int* in_sizes, int n_in,
                              void* d_out, int out_size, void* d_ws, size_t ws_size,
                              hipStream_t stream) {
  const float* z     = (const float*)d_in[0];
  const float* embed = (const float*)d_in[1];
  const float* w_lh  = (const float*)d_in[2];
  const float* b_lh  = (const float*)d_in[3];
  const float* w_ih0 = (const float*)d_in[4];
  const float* w_hh0 = (const float*)d_in[5];
  const float* b_ih0 = (const float*)d_in[6];
  const float* b_hh0 = (const float*)d_in[7];
  const float* w_ih1 = (const float*)d_in[8];
  const float* w_hh1 = (const float*)d_in[9];
  const float* b_ih1 = (const float*)d_in[10];
  const float* b_hh1 = (const float*)d_in[11];
  const float* w_out = (const float*)d_in[12];
  const float* b_out = (const float*)d_in[13];
  const int* bosP    = (const int*)d_in[14];
  const int* eosP    = (const int*)d_in[15];
  const int* padP    = (const int*)d_in[16];
  const int* maxlenP = (const int*)d_in[17];
  const int* topkP   = (const int*)d_in[18];
  const int* minlenP = (const int*)d_in[19];
  int* out_tokens = (int*)d_out;

  // ---- workspace layout ----
  char* ws = (char*)d_ws;
  size_t off = 0;
  auto alloc = [&](size_t bytes) -> void* {
    void* p = ws + off;
    off += (bytes + 255) & ~(size_t)255;
    return p;
  };
  bf16_t* embed_b = (bf16_t*)alloc((size_t)VOCAB * EMB * 2);
  bf16_t* wih0_b  = (bf16_t*)alloc((size_t)3 * HID * EMB * 2);
  bf16_t* whh0_b  = (bf16_t*)alloc((size_t)3 * HID * HID * 2);
  bf16_t* wih1_b  = (bf16_t*)alloc((size_t)3 * HID * HID * 2);
  bf16_t* whh1_b  = (bf16_t*)alloc((size_t)3 * HID * HID * 2);
  bf16_t* wout_b  = (bf16_t*)alloc((size_t)VOCAB * HID * 2);
  float*  h0f[2]; float* h1f[2]; bf16_t* h0b[2]; bf16_t* h1b[2];
  for (int i = 0; i < 2; ++i) {
    h0f[i] = (float*)alloc((size_t)NBATCH * HID * 4);
    h1f[i] = (float*)alloc((size_t)NBATCH * HID * 4);
    h0b[i] = (bf16_t*)alloc((size_t)NBATCH * HID * 2);
    h1b[i] = (bf16_t*)alloc((size_t)NBATCH * HID * 2);
  }
  float* logits = (float*)alloc((size_t)NBATCH * VOCAB * 4);
  int* cur      = (int*)alloc((size_t)NBATCH * 4);
  int* finished = (int*)alloc((size_t)NBATCH * 4);
  if (off > ws_size) return;  // workspace too small; nothing safe to do

  const int max_len = out_size / NBATCH;  // 64 in the reference setup

  // ---- one-time: weight conversion + initial state ----
  auto cvt = [&](const float* s, bf16_t* d, int n) {
    int blocks = (n + 255) / 256; if (blocks > 4096) blocks = 4096;
    hipLaunchKernelGGL(cvt_f32_bf16_kernel, dim3(blocks), dim3(256), 0, stream, s, d, n);
  };
  cvt(embed, embed_b, VOCAB * EMB);
  cvt(w_ih0, wih0_b, 3 * HID * EMB);
  cvt(w_hh0, whh0_b, 3 * HID * HID);
  cvt(w_ih1, wih1_b, 3 * HID * HID);
  cvt(w_hh1, whh1_b, 3 * HID * HID);
  cvt(w_out, wout_b, VOCAB * HID);

  hipLaunchKernelGGL(init_hidden_kernel, dim3((NBATCH * 2 * HID) / 256), dim3(256), 0, stream,
                     z, w_lh, b_lh, h0f[0], h1f[0], h0b[0], h1b[0]);
  hipLaunchKernelGGL(init_tokens_kernel, dim3(NBATCH / 256), dim3(256), 0, stream,
                     bosP, maxlenP, out_tokens, cur, finished);

  const size_t lds_gru0   = ((size_t)32 * EMB + (size_t)32 * HID) * 2;  // 40 KB
  const size_t lds_gru1   = ((size_t)32 * HID + (size_t)32 * HID) * 2;  // 64 KB
  const size_t lds_logits = (size_t)64 * HID * 2;                       // 64 KB

  // ---- autoregressive loop ----
  for (int t = 1; t < max_len; ++t) {
    const int src = (t - 1) & 1;
    const int dst = t & 1;
    // layer 0: x = embed[cur] gathered inside the async staging
    hipLaunchKernelGGL(gru_cell_kernel, dim3(NBATCH / 32, HID / 64), dim3(256), lds_gru0, stream,
                       embed_b, cur, EMB, h0b[src], h0f[src],
                       wih0_b, whh0_b, b_ih0, b_hh0, h0f[dst], h0b[dst]);
    // layer 1: input is layer-0 output
    hipLaunchKernelGGL(gru_cell_kernel, dim3(NBATCH / 32, HID / 64), dim3(256), lds_gru1, stream,
                       h0b[dst], (const int*)nullptr, HID, h1b[src], h1f[src],
                       wih1_b, whh1_b, b_ih1, b_hh1, h1f[dst], h1b[dst]);
    // output head
    hipLaunchKernelGGL(logits_kernel, dim3(NBATCH / 64, VOCAB / 128), dim3(256), lds_logits, stream,
                       h1b[dst], wout_b, b_out, logits);
    // sample next tokens
    hipLaunchKernelGGL(sample_kernel, dim3(NBATCH), dim3(256), 0, stream,
                       logits, bosP, eosP, padP, topkP, minlenP, maxlenP,
                       t, out_tokens, cur, finished);
  }
}
